// BERTInteractionHead_37589553774711
// MI455X (gfx1250) — matrix operand order repaired and checked by reading, not compiled, not run on hardware
//
#include <hip/hip_runtime.h>

// ---------------------------------------------------------------------------
// BERT cls-head forward, MI455X (gfx1250, wave32, WMMA + TDM).
// Only cls = (tanh(hidden[:,0] @ wp + bp) @ wm + bm) is required, so we only
// compute: full K/V projections (bf16 WMMA GEMMs, the dominant cost), Q at
// s=0, attention for q-row 0 per (b,h), and 16-row GEMMs for the rest.
// The big GEMM stages tiles into LDS with tensor_load_to_lds (TDM, TENSORcnt)
// and double-buffers the DMA behind the WMMA stream.
// ---------------------------------------------------------------------------

typedef __bf16 bf16;
typedef __attribute__((ext_vector_type(16))) __bf16 bf16x16;
typedef __attribute__((ext_vector_type(8)))  __bf16 bf16x8;
typedef __attribute__((ext_vector_type(8)))  float  f32x8;
typedef __attribute__((ext_vector_type(4)))  unsigned int u32x4;
typedef __attribute__((ext_vector_type(8)))  int i32x8;
typedef __attribute__((ext_vector_type(4)))  int i32x4;

union FragAB { bf16x16 v; bf16x8 h[2]; };

constexpr int Bb = 16, Ss = 1024, Hh = 768, NHn = 12, FFf = 3072, Dd = 64;
constexpr int BS = Bb * Ss; // 16384 rows

// ---- bf16 helpers (round-to-nearest-even) ---------------------------------
__device__ __forceinline__ bf16 f2b(float f) {
    union { float f; unsigned u; } x; x.f = f;
    unsigned r = x.u + 0x7FFFu + ((x.u >> 16) & 1u);
    union { unsigned short s; bf16 b; } o; o.s = (unsigned short)(r >> 16);
    return o.b;
}
__device__ __forceinline__ float b2f(bf16 b) {
    union { unsigned short s; bf16 b; } i; i.b = b;
    union { unsigned u; float f; } o; o.u = ((unsigned)i.s) << 16;
    return o.f;
}

__device__ __forceinline__ f32x8 wmma_bf16(bf16x16 a, bf16x16 b, f32x8 c) {
    return __builtin_amdgcn_wmma_f32_16x16x32_bf16(
        /*neg_a=*/false, a, /*neg_b=*/false, b,
        /*c_mod=*/(short)0, c, /*reuse_a=*/false, /*reuse_b=*/false);
}

// Low 32 bits of a generic pointer into LDS = byte offset within LDS.
__device__ __forceinline__ unsigned lds_off(const void* p) {
    return (unsigned)(unsigned long long)(uintptr_t)p;
}

// ---------------------------------------------------------------------------
// TDM: issue a 2D tile load (tileW dwords x tileH rows) from a row-major
// dword tensor (row stride = rowDw dwords, rows = tensor rows) into LDS at
// byte offset ldsAddr. D# fields per CDNA5 ISA 8.3/8.4 (count=1, type=2,
// data_size=4B, no padding / iterate / multicast). Groups 2,3 zero (2D).
// This toolchain exposes the 6-arg builtin form (extra i32x8 before cpol).
// ---------------------------------------------------------------------------
__device__ __forceinline__ void tdm_load_2d(unsigned ldsAddr, const void* gptr,
                                            unsigned rowDw, unsigned rows,
                                            unsigned tileW, unsigned tileH) {
    unsigned long long ga = (unsigned long long)(uintptr_t)gptr;
    u32x4 g0;
    g0[0] = 1u;                                   // count=1 (valid descriptor)
    g0[1] = ldsAddr;                              // lds_addr (bytes)
    g0[2] = (unsigned)(ga & 0xFFFFFFFFull);       // global_addr[31:0]
    g0[3] = (unsigned)((ga >> 32) & 0x01FFFFFFull) | (2u << 30); // addr[56:32] | type=2

    i32x8 g1;
    g1[0] = (int)(2u << 16);                                        // data_size=4B
    g1[1] = (int)((rowDw & 0xFFFFu) << 16);                         // tensor_dim0[15:0]
    g1[2] = (int)(((rowDw >> 16) & 0xFFFFu) | ((rows & 0xFFFFu) << 16));  // dim0[31:16]|dim1[15:0]
    g1[3] = (int)(((rows >> 16) & 0xFFFFu) | ((tileW & 0xFFFFu) << 16)); // dim1[31:16]|tile_dim0
    g1[4] = (int)(tileH & 0xFFFFu);                                 // tile_dim1 | tile_dim2=0
    g1[5] = (int)rowDw;                                             // tensor_dim0_stride[31:0]
    g1[6] = 0;                                                      // stride0[47:32]|stride1[15:0]
    g1[7] = 0;                                                      // stride1[47:16]

    i32x4 z4; z4[0] = 0; z4[1] = 0; z4[2] = 0; z4[3] = 0;
    i32x8 z8; z8[0] = 0; z8[1] = 0; z8[2] = 0; z8[3] = 0;
    z8[4] = 0; z8[5] = 0; z8[6] = 0; z8[7] = 0;
    __builtin_amdgcn_tensor_load_to_lds(g0, g1, z4, z4, z8, 0);
}

// ---------------------------------------------------------------------------
// fp32 -> bf16 bulk convert (4 elems / thread)
// ---------------------------------------------------------------------------
__global__ __launch_bounds__(256) void cvt4_kernel(const float* __restrict__ in,
                                                   bf16* __restrict__ out, int n4) {
    int i = blockIdx.x * 256 + threadIdx.x;
    if (i >= n4) return;
    float4 v = ((const float4*)in)[i];
    union { bf16 b[4]; unsigned long long u; } p;
    p.b[0] = f2b(v.x); p.b[1] = f2b(v.y); p.b[2] = f2b(v.z); p.b[3] = f2b(v.w);
    ((unsigned long long*)out)[i] = p.u;
}

// ---------------------------------------------------------------------------
// Transposed fp32->bf16 weight copy: in[K][N] -> out[N][K]
// so every WMMA B-fragment is a contiguous 32-byte read.
// dims multiples of 32. block (32,8).
// ---------------------------------------------------------------------------
__global__ __launch_bounds__(256) void transpose_cvt_kernel(const float* __restrict__ in,
                                                            bf16* __restrict__ out,
                                                            int K, int N) {
    __shared__ float tile[32][33];
    int nx = blockIdx.x * 32, kx = blockIdx.y * 32;
    for (int i = threadIdx.y; i < 32; i += 8)
        tile[i][threadIdx.x] = in[(size_t)(kx + i) * N + nx + threadIdx.x];
    __syncthreads();
    for (int i = threadIdx.y; i < 32; i += 8)
        out[(size_t)(nx + i) * K + kx + threadIdx.x] = f2b(tile[threadIdx.x][i]);
}

// ---------------------------------------------------------------------------
// Big bf16 WMMA GEMM:  C[M][N] = A[M][K] @ Bt[N][K]^T + bias,  bf16 output.
// BM=128 BN=128 BK=32; 8 waves, each wave computes 32x64 (2x4 16x16 tiles).
// LDS tiles are staged by the Tensor Data Mover (wave 0 issues 2 descriptors
// per tile, double-buffered; TENSORcnt completes in order so
// s_wait_tensorcnt(2) guarantees the previous tile is resident).
// M,N multiples of 128; K multiple of 32 (16384 x 768 x 768 here).
// ---------------------------------------------------------------------------
#define BM 128
#define BN 128
#define BK 32
__global__ __launch_bounds__(256) void gemm_bf16_wmma_kernel(
    const bf16* __restrict__ A, const bf16* __restrict__ Bt,
    const float* __restrict__ bias, bf16* __restrict__ C,
    int M, int N, int K)
{
    __shared__ __align__(64) bf16 As[2][BM * BK];
    __shared__ __align__(64) bf16 Bs[2][BN * BK];

    const int tid  = threadIdx.x;
    const int lane = tid & 31;
    const int wid  = tid >> 5;
    const int waveM = wid & 3;   // 4 waves along M (32 rows each)
    const int waveN = wid >> 2;  // 2 waves along N (64 cols each)
    const int half = lane >> 4;
    const int l15  = lane & 15;

    const int m0 = blockIdx.y * BM;
    const int n0 = blockIdx.x * BN;

    const unsigned rowDw = (unsigned)(K >> 1);   // K bf16 = K/2 dwords per row
    const unsigned asOff0 = lds_off(&As[0][0]), asOff1 = lds_off(&As[1][0]);
    const unsigned bsOff0 = lds_off(&Bs[0][0]), bsOff1 = lds_off(&Bs[1][0]);

    f32x8 acc[2][4];
    const f32x8 z = {0.f, 0.f, 0.f, 0.f, 0.f, 0.f, 0.f, 0.f};
#pragma unroll
    for (int mt = 0; mt < 2; ++mt)
#pragma unroll
        for (int nt = 0; nt < 4; ++nt) acc[mt][nt] = z;

    const int nk = K / BK;
    if (wid == 0) { // preload tile 0 via TDM
        tdm_load_2d(asOff0, A  + (size_t)m0 * K, rowDw, (unsigned)M, BK / 2, BM);
        tdm_load_2d(bsOff0, Bt + (size_t)n0 * K, rowDw, (unsigned)N, BK / 2, BN);
    }

    for (int kt = 0; kt < nk; ++kt) {
        const int buf = kt & 1;
        if (wid == 0) {
            if (kt + 1 < nk) { // stream next tile into the other buffer
                tdm_load_2d(buf ? asOff0 : asOff1,
                            A + (size_t)m0 * K + (size_t)(kt + 1) * BK,
                            rowDw, (unsigned)M, BK / 2, BM);
                tdm_load_2d(buf ? bsOff0 : bsOff1,
                            Bt + (size_t)n0 * K + (size_t)(kt + 1) * BK,
                            rowDw, (unsigned)N, BK / 2, BN);
                __builtin_amdgcn_s_wait_tensorcnt(2); // tile kt done (in-order)
            } else {
                __builtin_amdgcn_s_wait_tensorcnt(0);
            }
        }
        __syncthreads(); // publish tile kt to all waves

        const bf16* as = &As[buf][0];
        const bf16* bs = &Bs[buf][0];

        // A fragments: lane<16 -> M=lane, K=0..15; lane>=16 -> K=16..31 (contiguous)
        FragAB af[2];
#pragma unroll
        for (int mt = 0; mt < 2; ++mt) {
            int r    = waveM * 32 + mt * 16 + l15;
            int base = r * BK + (half ? 16 : 0);
            af[mt].h[0] = *(const bf16x8*)&as[base];
            af[mt].h[1] = *(const bf16x8*)&as[base + 8];
        }
        // B fragments: lane<16 -> N=lane, K=0..15; lane>=16 -> K=16..31 (contiguous)
        bf16x16 bfg[4];
#pragma unroll
        for (int nt = 0; nt < 4; ++nt) {
            int n    = waveN * 64 + nt * 16 + l15;
            int base = n * BK + (half ? 16 : 0);
            bfg[nt] = *(const bf16x16*)&bs[base];
        }
#pragma unroll
        for (int mt = 0; mt < 2; ++mt)
#pragma unroll
            for (int nt = 0; nt < 4; ++nt)
                acc[mt][nt] = wmma_bf16(af[mt].v, bfg[nt], acc[mt][nt]);

        __syncthreads(); // all waves done with buf before TDM overwrites it
    }

    // epilogue: bias + bf16 store. C tile: lane=N, VGPR r -> M=r(+8 if lane>=16)
#pragma unroll
    for (int mt = 0; mt < 2; ++mt) {
#pragma unroll
        for (int nt = 0; nt < 4; ++nt) {
            int n  = n0 + waveN * 64 + nt * 16 + l15;
            float bv = bias[n];
            int rbase = m0 + waveM * 32 + mt * 16 + (half ? 8 : 0);
#pragma unroll
            for (int r = 0; r < 8; ++r)
                C[(size_t)(rbase + r) * N + n] = f2b(acc[mt][nt][r] + bv);
        }
    }
}

// ---------------------------------------------------------------------------
// 16-row WMMA GEMM with strided A-row gather + fused epilogues.
//   C[16][N] = gather16(A, aRowStride) @ Bt[N][K]^T + bias, then:
//   mode 0: store bf16
//   mode 1: += resid[row*residStride + n], store f32
//   mode 2: GELU(exact), store bf16
//   mode 3: tanh, store f32
// 8 waves -> 128 cols per block; grid.x = N/128. K multiple of 32.
// ---------------------------------------------------------------------------
__global__ __launch_bounds__(256) void gemm16_wmma_kernel(
    const bf16* __restrict__ A, long aRowStride,
    const bf16* __restrict__ Bt, const float* __restrict__ bias,
    void* __restrict__ Cout,
    const float* __restrict__ resid, long residStride,
    int N, int K, int mode)
{
    __shared__ __align__(64) bf16 As[16 * 32];
    const int tid  = threadIdx.x;
    const int lane = tid & 31;
    const int wid  = tid >> 5;
    const int half = lane >> 4;
    const int l15  = lane & 15;
    const int n0   = blockIdx.x * 128 + wid * 16;

    f32x8 acc = {0.f, 0.f, 0.f, 0.f, 0.f, 0.f, 0.f, 0.f};

    for (int k0 = 0; k0 < K; k0 += 32) {
        if (tid < 64) { // stage 16x32 A tile (gathered rows)
            int row = tid >> 2;
            int col = (tid & 3) << 3;
            *(bf16x8*)&As[row * 32 + col] =
                *(const bf16x8*)&A[(size_t)row * aRowStride + k0 + col];
        }
        __syncthreads();

        FragAB af;
        {
            int base = l15 * 32 + (half ? 16 : 0);
            af.h[0] = *(const bf16x8*)&As[base];
            af.h[1] = *(const bf16x8*)&As[base + 8];
        }
        // B fragment straight from global (contiguous 32B per lane, no reuse)
        bf16x16 bfg = *(const bf16x16*)&Bt[(size_t)(n0 + l15) * K + k0 + (half ? 16 : 0)];

        acc = wmma_bf16(af.v, bfg, acc);
        __syncthreads();
    }

    const int n = n0 + l15;
    const float bv = bias[n];
    const int rbase = half ? 8 : 0;
#pragma unroll
    for (int r = 0; r < 8; ++r) {
        int row = rbase + r;
        float v = acc[r] + bv;
        if (mode == 0) {
            ((bf16*)Cout)[(size_t)row * N + n] = f2b(v);
        } else if (mode == 1) {
            ((float*)Cout)[(size_t)row * N + n] = v + resid[(size_t)row * residStride + n];
        } else if (mode == 2) {
            float g = 0.5f * v * (1.0f + erff(v * 0.70710678118654752f));
            ((bf16*)Cout)[(size_t)row * N + n] = f2b(g);
        } else {
            ((float*)Cout)[(size_t)row * N + n] = tanhf(v);
        }
    }
}

// ---------------------------------------------------------------------------
// Attention for q-position 0, one block per (b, h). Streams K/V (bf16).
// ---------------------------------------------------------------------------
__global__ __launch_bounds__(256) void attn_cls_kernel(
    const bf16* __restrict__ Q0,   // [16][768], row=b, col=h*64+d
    const bf16* __restrict__ Kb,   // [B*S][768]
    const bf16* __restrict__ Vb,   // [B*S][768]
    const float* __restrict__ mask,// [B][S]
    bf16* __restrict__ ctx0)       // [16][768]
{
    const int b = blockIdx.x / NHn;
    const int h = blockIdx.x % NHn;
    const int tid = threadIdx.x;

    __shared__ float q[Dd];
    __shared__ float sc[Ss];
    __shared__ float red[256];

    if (tid < Dd) q[tid] = b2f(Q0[(size_t)b * Hh + h * Dd + tid]);
    __syncthreads();

    const bf16* Kh = Kb + (size_t)b * Ss * Hh + h * Dd;
    float lmax = -3.0e38f;
    for (int s = tid; s < Ss; s += 256) {
        const bf16* kr = Kh + (size_t)s * Hh;
        __builtin_prefetch(kr + (size_t)256 * Hh, 0, 1);
        float a = 0.f;
#pragma unroll
        for (int d = 0; d < Dd; ++d) a += q[d] * b2f(kr[d]);
        a = a * 0.125f + mask[(size_t)b * Ss + s]; // 1/sqrt(64)
        sc[s] = a;
        lmax = fmaxf(lmax, a);
    }
    red[tid] = lmax; __syncthreads();
    for (int o = 128; o > 0; o >>= 1) {
        if (tid < o) red[tid] = fmaxf(red[tid], red[tid + o]);
        __syncthreads();
    }
    float mx = red[0];
    __syncthreads();

    float lsum = 0.f;
    for (int s = tid; s < Ss; s += 256) {
        float p = __expf(sc[s] - mx);
        sc[s] = p;
        lsum += p;
    }
    red[tid] = lsum; __syncthreads();
    for (int o = 128; o > 0; o >>= 1) {
        if (tid < o) red[tid] += red[tid + o];
        __syncthreads();
    }
    float inv = 1.0f / red[0];
    __syncthreads();

    // ctx[d] = sum_s p[s] * V[s][d]; 4 s-groups x 64 d-lanes (coalesced rows)
    const bf16* Vh = Vb + (size_t)b * Ss * Hh + h * Dd;
    const int d = tid & 63, grp = tid >> 6;
    float av = 0.f;
    for (int s = grp; s < Ss; s += 4) av += sc[s] * b2f(Vh[(size_t)s * Hh + d]);
    red[tid] = av; __syncthreads();
    if (grp == 0) {
        float v = red[d] + red[64 + d] + red[128 + d] + red[192 + d];
        ctx0[(size_t)b * Hh + h * Dd + d] = f2b(v * inv);
    }
}

// ---------------------------------------------------------------------------
// 16-row LayerNorm: one block per row. Writes f32 (for residual) + bf16.
// ---------------------------------------------------------------------------
__global__ __launch_bounds__(256) void ln16_kernel(
    const float* __restrict__ X, const float* __restrict__ g,
    const float* __restrict__ be, float* __restrict__ Yf,
    bf16* __restrict__ Yb, int N)
{
    const int row = blockIdx.x;
    const int tid = threadIdx.x;
    __shared__ float red[256];
    const float* x = X + (size_t)row * N;

    float s = 0.f;
    for (int i = tid; i < N; i += 256) s += x[i];
    red[tid] = s; __syncthreads();
    for (int o = 128; o > 0; o >>= 1) { if (tid < o) red[tid] += red[tid + o]; __syncthreads(); }
    float mean = red[0] / (float)N;
    __syncthreads();

    float v = 0.f;
    for (int i = tid; i < N; i += 256) { float d = x[i] - mean; v += d * d; }
    red[tid] = v; __syncthreads();
    for (int o = 128; o > 0; o >>= 1) { if (tid < o) red[tid] += red[tid + o]; __syncthreads(); }
    float rstd = rsqrtf(red[0] / (float)N + 1e-12f);

    for (int i = tid; i < N; i += 256) {
        float y = (x[i] - mean) * rstd * g[i] + be[i];
        Yf[(size_t)row * N + i] = y;
        Yb[(size_t)row * N + i] = f2b(y);
    }
}

// ---------------------------------------------------------------------------
// cls[b] = sum_j pooled[b][j] * wm[j] + bm ; one block per b
// ---------------------------------------------------------------------------
__global__ __launch_bounds__(256) void cls_kernel(
    const float* __restrict__ pooled, const float* __restrict__ wm,
    const float* __restrict__ bm, float* __restrict__ out, int N)
{
    __shared__ float red[256];
    const int b = blockIdx.x, tid = threadIdx.x;
    float s = 0.f;
    for (int j = tid; j < N; j += 256) s += pooled[(size_t)b * N + j] * wm[j];
    red[tid] = s; __syncthreads();
    for (int o = 128; o > 0; o >>= 1) { if (tid < o) red[tid] += red[tid + o]; __syncthreads(); }
    if (tid == 0) out[b] = red[0] + bm[0];
}

// ---------------------------------------------------------------------------
extern "C" void kernel_launch(void* const* d_in, const int* in_sizes, int n_in,
                              void* d_out, int out_size, void* d_ws, size_t ws_size,
                              hipStream_t stream)
{
    (void)in_sizes; (void)n_in; (void)out_size; (void)ws_size;

    const float* features = (const float*)d_in[0];
    const float* mask     = (const float*)d_in[1];
    const float* wq = (const float*)d_in[2];  const float* bq = (const float*)d_in[3];
    const float* wk = (const float*)d_in[4];  const float* bk = (const float*)d_in[5];
    const float* wv = (const float*)d_in[6];  const float* bv = (const float*)d_in[7];
    const float* wo = (const float*)d_in[8];  const float* bo = (const float*)d_in[9];
    const float* ln1_g = (const float*)d_in[10]; const float* ln1_b = (const float*)d_in[11];
    const float* w1 = (const float*)d_in[12]; const float* b1 = (const float*)d_in[13];
    const float* w2 = (const float*)d_in[14]; const float* b2 = (const float*)d_in[15];
    const float* ln2_g = (const float*)d_in[16]; const float* ln2_b = (const float*)d_in[17];
    const float* wp = (const float*)d_in[18]; const float* bp = (const float*)d_in[19];
    const float* wm = (const float*)d_in[20]; const float* bm = (const float*)d_in[21];
    float* out = (float*)d_out;

    // ---- workspace carve-up (all 256B aligned) ----
    char* ws = (char*)d_ws;
    size_t off = 0;
    auto alloc = [&](size_t bytes) { size_t o = off; off = (off + bytes + 255) & ~(size_t)255; return o; };

    bf16* Xbf  = (bf16*)(ws + alloc((size_t)BS * Hh * 2));
    bf16* WqT  = (bf16*)(ws + alloc((size_t)Hh * Hh * 2));
    bf16* WkT  = (bf16*)(ws + alloc((size_t)Hh * Hh * 2));
    bf16* WvT  = (bf16*)(ws + alloc((size_t)Hh * Hh * 2));
    bf16* WoT  = (bf16*)(ws + alloc((size_t)Hh * Hh * 2));
    bf16* WpT  = (bf16*)(ws + alloc((size_t)Hh * Hh * 2));
    bf16* W1T  = (bf16*)(ws + alloc((size_t)Hh * FFf * 2)); // [3072][768]
    bf16* W2T  = (bf16*)(ws + alloc((size_t)FFf * Hh * 2)); // [768][3072]
    bf16* Kbf  = (bf16*)(ws + alloc((size_t)BS * Hh * 2));
    bf16* Vbf  = (bf16*)(ws + alloc((size_t)BS * Hh * 2));
    bf16* Q0   = (bf16*)(ws + alloc((size_t)Bb * Hh * 2));
    bf16* CTX0 = (bf16*)(ws + alloc((size_t)Bb * Hh * 2));
    float* ATT0   = (float*)(ws + alloc((size_t)Bb * Hh * 4)); // ctx@wo+bo+resid
    float* ATT0F  = (float*)(ws + alloc((size_t)Bb * Hh * 4)); // LN1 out f32
    bf16*  ATT0B  = (bf16*)(ws + alloc((size_t)Bb * Hh * 2));  // LN1 out bf16
    bf16*  FFN1   = (bf16*)(ws + alloc((size_t)Bb * FFf * 2)); // gelu out
    float* FFN2   = (float*)(ws + alloc((size_t)Bb * Hh * 4)); // +resid
    float* HIDF   = (float*)(ws + alloc((size_t)Bb * Hh * 4));
    bf16*  HIDB   = (bf16*)(ws + alloc((size_t)Bb * Hh * 2));
    float* POOL   = (float*)(ws + alloc((size_t)Bb * Hh * 4));

    // 1) X -> bf16
    {
        int n4 = (BS * Hh) / 4;
        cvt4_kernel<<<(n4 + 255) / 256, 256, 0, stream>>>(features, Xbf, n4);
    }
    // 2) transposed bf16 weights
    {
        dim3 blk(32, 8);
        transpose_cvt_kernel<<<dim3(Hh / 32, Hh / 32), blk, 0, stream>>>(wq, WqT, Hh, Hh);
        transpose_cvt_kernel<<<dim3(Hh / 32, Hh / 32), blk, 0, stream>>>(wk, WkT, Hh, Hh);
        transpose_cvt_kernel<<<dim3(Hh / 32, Hh / 32), blk, 0, stream>>>(wv, WvT, Hh, Hh);
        transpose_cvt_kernel<<<dim3(Hh / 32, Hh / 32), blk, 0, stream>>>(wo, WoT, Hh, Hh);
        transpose_cvt_kernel<<<dim3(Hh / 32, Hh / 32), blk, 0, stream>>>(wp, WpT, Hh, Hh);
        transpose_cvt_kernel<<<dim3(FFf / 32, Hh / 32), blk, 0, stream>>>(w1, W1T, Hh, FFf);
        transpose_cvt_kernel<<<dim3(Hh / 32, FFf / 32), blk, 0, stream>>>(w2, W2T, FFf, Hh);
    }
    // 3) K = X@Wk + bk, V = X@Wv + bv  (the two big TDM+WMMA GEMMs)
    {
        dim3 grid(Hh / BN, BS / BM); // (6, 128)
        gemm_bf16_wmma_kernel<<<grid, 256, 0, stream>>>(Xbf, WkT, bk, Kbf, BS, Hh, Hh);
        gemm_bf16_wmma_kernel<<<grid, 256, 0, stream>>>(Xbf, WvT, bv, Vbf, BS, Hh, Hh);
    }
    // 4) Q0 = X[s=0 rows] @ Wq + bq  (16-row WMMA, rows gathered at stride S*H)
    gemm16_wmma_kernel<<<Hh / 128, 256, 0, stream>>>(
        Xbf, (long)Ss * Hh, WqT, bq, (void*)Q0, nullptr, 0, Hh, Hh, 0);
    // 5) attention at q-position 0
    attn_cls_kernel<<<Bb * NHn, 256, 0, stream>>>(Q0, Kbf, Vbf, mask, CTX0);
    // 6) ATT0 = CTX0 @ Wo + bo + features[s=0]   (f32)
    gemm16_wmma_kernel<<<Hh / 128, 256, 0, stream>>>(
        CTX0, (long)Hh, WoT, bo, (void*)ATT0, features, (long)Ss * Hh, Hh, Hh, 1);
    // 7) LN1
    ln16_kernel<<<Bb, 256, 0, stream>>>(ATT0, ln1_g, ln1_b, ATT0F, ATT0B, Hh);
    // 8) FFN1 = gelu(ATT0B @ W1 + b1)  (bf16)
    gemm16_wmma_kernel<<<FFf / 128, 256, 0, stream>>>(
        ATT0B, (long)Hh, W1T, b1, (void*)FFN1, nullptr, 0, FFf, Hh, 2);
    // 9) FFN2 = FFN1 @ W2 + b2 + ATT0F  (f32)
    gemm16_wmma_kernel<<<Hh / 128, 256, 0, stream>>>(
        FFN1, (long)FFf, W2T, b2, (void*)FFN2, ATT0F, (long)Hh, Hh, FFf, 1);
    // 10) LN2
    ln16_kernel<<<Bb, 256, 0, stream>>>(FFN2, ln2_g, ln2_b, HIDF, HIDB, Hh);
    // 11) pooled = tanh(HIDB @ Wp + bp)  (f32)
    gemm16_wmma_kernel<<<Hh / 128, 256, 0, stream>>>(
        HIDB, (long)Hh, WpT, bp, (void*)POOL, nullptr, 0, Hh, Hh, 3);
    // 12) cls = pooled @ wm + bm
    cls_kernel<<<Bb, 256, 0, stream>>>(POOL, wm, bm, out, Hh);
}
